// AutoRegressive_67491116089932
// MI455X (gfx1250) — compile-verified, matrix operand order checked by reading, not run on hardware
//
#include <hip/hip_runtime.h>
#include <hip/hip_bf16.h>

typedef __attribute__((ext_vector_type(16))) _Float16 v16h;
typedef __attribute__((ext_vector_type(8)))  _Float16 v8h;
typedef __attribute__((ext_vector_type(8)))  float    v8f;

#define NB     256   // batch
#define TWC    512   // TW == TC == 512
#define NFEAT  12    // F + C
#define NK     16    // time2vec dim
#define NH     64    // hidden
#define INSZ   36    // 2F + C + K
#define OUTSZ  16    // 2F
#define KPAD   128   // padded K for combined [input | h] matmul
#define NGATE  256   // 4 * H
#define BT     16    // batch tile per workgroup
#define THREADS 256
#define NWAVE  8

// Branchless fast activations on native v_exp_f32 / v_rcp_f32 paths.
__device__ __forceinline__ float sigmf(float v) {
  return __builtin_amdgcn_rcpf(1.f + __expf(-v));
}
__device__ __forceinline__ float tanh_fast(float v) {
  // 1 - 2/(e^{2v}+1); saturates to +/-1 for large |v| without branches.
  return 1.f - 2.f * __builtin_amdgcn_rcpf(__expf(2.f * v) + 1.f);
}

// Load a 16x32 f16 A-fragment from LDS row-major A[BT][KPAD] per CDNA5 layout:
// lane l (l<16): M=l, halves 0..7 -> K=kc*32+0..7, halves 8..15 -> K=kc*32+16..23
// lane l (>=16): M=l-16, same but K offset +8. Two contiguous 16B LDS loads.
__device__ __forceinline__ v16h load_a_frag(const _Float16* Abase, int M, int kc, int lhi) {
  const v8h* p = (const v8h*)(Abase + M * KPAD + kc * 32 + 8 * lhi);
  v8h lo = p[0];   // K = base + 0..7
  v8h hi = p[2];   // K = base + 16..23
  v16h r;
#pragma unroll
  for (int i = 0; i < 8; ++i) { r[i] = lo[i]; r[8 + i] = hi[i]; }
  return r;
}

// One layer's gate GEMM: gates(16,256) = A(16,128) @ W^T + bias, via WMMA.
// Each wave owns N-tiles {2*wave, 2*wave+1}; 4 K-chunks of 32.
__device__ __forceinline__ void gemm_layer(const _Float16* Abase, float (*gates)[NGATE],
                                           const float* biasL, const v16h (&wf)[2][4],
                                           int wave, int lmod, int lhi) {
  const int n0 = (wave * 2 + 0) * 16 + lmod;
  const int n1 = n0 + 16;
  const float b0 = biasL[n0], b1 = biasL[n1];
  v8f acc0, acc1;
#pragma unroll
  for (int r = 0; r < 8; ++r) { acc0[r] = b0; acc1[r] = b1; }
#pragma unroll
  for (int kc = 0; kc < 4; ++kc) {
    v16h a = load_a_frag(Abase, lmod, kc, lhi);
    acc0 = __builtin_amdgcn_wmma_f32_16x16x32_f16(false, a, false, wf[0][kc], (short)0, acc0, false, false);
    acc1 = __builtin_amdgcn_wmma_f32_16x16x32_f16(false, a, false, wf[1][kc], (short)0, acc1, false, false);
  }
#pragma unroll
  for (int r = 0; r < 8; ++r) {
    gates[r + 8 * lhi][n0] = acc0[r];
    gates[r + 8 * lhi][n1] = acc1[r];
  }
}

__global__ void zero_kernel(float* p, int n) {
  int i = blockIdx.x * blockDim.x + threadIdx.x;
  if (i < n) p[i] = 0.f;
}

__global__ __launch_bounds__(THREADS) void ar_lstm_kernel(
    const float* __restrict__ x, const int* __restrict__ x_lens,
    const float* __restrict__ time_, const float* __restrict__ context,
    const int* __restrict__ ctx_lens, const float* __restrict__ omega,
    const float* __restrict__ phi,
    const float* __restrict__ Wih0, const float* __restrict__ Whh0,
    const float* __restrict__ bih0, const float* __restrict__ bhh0,
    const float* __restrict__ Wih1, const float* __restrict__ Whh1,
    const float* __restrict__ bih1, const float* __restrict__ bhh1,
    const float* __restrict__ Wd, const float* __restrict__ bd,
    float* __restrict__ out) {
  __shared__ __align__(16) _Float16 A[BT * KPAD];       // WMMA A staging (16 x 128)
  __shared__ float    gates[BT][NGATE];                 // WMMA D staging (f32)
  __shared__ _Float16 hbuf[2][BT][NH];                  // live h (f16 for matmul)
  __shared__ float    cbuf[2][BT][NH];                  // live c (f32)
  __shared__ _Float16 sh[2][BT][NH];                    // snapshot h at x_lens-1
  __shared__ float    sc[2][BT][NH];                    // snapshot c
  __shared__ float    outst[BT][OUTSZ];                 // projection output staging
  __shared__ _Float16 elem[BT][OUTSZ];                  // warmup prediction (f16)
  __shared__ float    biasL[2][NGATE];
  __shared__ float    om[NK], ph[NK], bdl[OUTSZ];
  __shared__ int      xl[BT], cl[BT];

  const int tid  = threadIdx.x;
  const int wave = tid >> 5;
  const int lane = tid & 31;
  const int lmod = lane & 15;
  const int lhi  = lane >> 4;
  const int b0   = blockIdx.x * BT;

  // ---- one-time init ----
  for (int n = tid; n < NGATE; n += THREADS) {
    biasL[0][n] = bih0[n] + bhh0[n];
    biasL[1][n] = bih1[n] + bhh1[n];
  }
  if (tid < NK) { om[tid] = omega[tid]; ph[tid] = phi[tid]; }
  if (tid < OUTSZ) bdl[tid] = bd[tid];
  if (tid < BT) { xl[tid] = x_lens[b0 + tid]; cl[tid] = ctx_lens[b0 + tid]; }
  for (int i = tid; i < 2 * BT * NH; i += THREADS) {
    ((_Float16*)hbuf)[i] = (_Float16)0.f;
    ((float*)cbuf)[i]    = 0.f;
  }

  // ---- weights -> per-wave register fragments (B-matrix layout, mirrors A) ----
  // Combined K layout, layer0: [x_in(36) | pad->64 | h0(64)]; layer1: [h0(64) | h1(64)].
  v16h wfrag[2][2][4];
#pragma unroll
  for (int layer = 0; layer < 2; ++layer) {
#pragma unroll
    for (int ti = 0; ti < 2; ++ti) {
      const int n = (wave * 2 + ti) * 16 + lmod;   // gate column (N)
#pragma unroll
      for (int kc = 0; kc < 4; ++kc) {
        v16h f;
#pragma unroll
        for (int h8 = 0; h8 < 16; ++h8) {
          const int kk = kc * 32 + (h8 < 8 ? h8 : h8 + 8) + 8 * lhi;
          float v;
          if (layer == 0)
            v = (kk < INSZ) ? Wih0[n * INSZ + kk] : (kk < NH ? 0.f : Whh0[n * NH + (kk - NH)]);
          else
            v = (kk < NH) ? Wih1[n * NH + kk] : Whh1[n * NH + (kk - NH)];
          f[h8] = (_Float16)v;
        }
        wfrag[layer][ti][kc] = f;
      }
    }
  }
  // Output projection Wd (16 x 64): single N-tile, 2 K-chunks (used by wave 0).
  v16h wdf[2];
#pragma unroll
  for (int kc = 0; kc < 2; ++kc) {
    v16h f;
#pragma unroll
    for (int h8 = 0; h8 < 16; ++h8) {
      const int kk = kc * 32 + (h8 < 8 ? h8 : h8 + 8) + 8 * lhi;
      f[h8] = (_Float16)Wd[lmod * NH + kk];
    }
    wdf[kc] = f;
  }
  __syncthreads();

  // Shared activation step: gates+c -> (h,c); optional writes into A and snapshots.
  auto activate = [&](int layer, bool toA_lo, bool toA_hi_h1, int snap_t) {
#pragma unroll
    for (int q = 0; q < 4; ++q) {
      const int idx = tid * 4 + q;
      const int row = idx >> 6, j = idx & 63;
      const float cold = cbuf[layer][row][j];
      const float cn = sigmf(gates[row][64 + j]) * cold +
                       sigmf(gates[row][j]) * tanh_fast(gates[row][128 + j]);
      const float hn = sigmf(gates[row][192 + j]) * tanh_fast(cn);
      cbuf[layer][row][j] = cn;
      const _Float16 hh = (_Float16)hn;
      hbuf[layer][row][j] = hh;
      if (toA_lo)    A[row * KPAD + j] = hh;                       // next input, cols 0..63
      if (toA_hi_h1) A[row * KPAD + 64 + j] = hbuf[1][row][j];     // old h1, cols 64..127
      if (snap_t >= 0 && snap_t == xl[row] - 1) { sh[layer][row][j] = hh; sc[layer][row][j] = cn; }
    }
  };

  auto project = [&]() {  // outst = A[:, 0:64] @ Wd^T + bd   (wave 0 only)
    if (wave == 0) {
      v8f acc;
      const float bv = bdl[lmod];
#pragma unroll
      for (int r = 0; r < 8; ++r) acc[r] = bv;
#pragma unroll
      for (int kc = 0; kc < 2; ++kc) {
        v16h a = load_a_frag(A, lmod, kc, lhi);
        acc = __builtin_amdgcn_wmma_f32_16x16x32_f16(false, a, false, wdf[kc], (short)0, acc, false, false);
      }
#pragma unroll
      for (int r = 0; r < 8; ++r) outst[r + 8 * lhi][lmod] = acc[r];
    }
  };

  // ================= warmup over TW steps =================
  for (int t = 0; t < TWC; ++t) {
    for (int idx = tid; idx < BT * KPAD; idx += THREADS) {
      const int row = idx >> 7, col = idx & 127;
      _Float16 v;
      if (col < NFEAT)      v = (_Float16)x[((b0 + row) * TWC + t) * NFEAT + col];
      else if (col < 20)    v = (_Float16)1.f;  // std placeholder
      else if (col < 36)    v = (_Float16)__sinf(time_[(b0 + row) * TWC + t] * om[col - 20] + ph[col - 20]);
      else if (col < 64)    v = (_Float16)0.f;
      else                  v = hbuf[0][row][col - 64];
      A[idx] = v;
    }
    __syncthreads();
    gemm_layer(A, gates, biasL[0], wfrag[0], wave, lmod, lhi);
    __syncthreads();
    activate(0, /*toA_lo=*/true, /*toA_hi_h1=*/true, /*snap_t=*/t);
    __syncthreads();
    gemm_layer(A, gates, biasL[1], wfrag[1], wave, lmod, lhi);
    __syncthreads();
    activate(1, false, false, t);
    __syncthreads();
  }

  // ================= element = h1(saved) @ Wd^T + bd =================
  for (int i = tid; i < 2 * BT * NH; i += THREADS) {
    ((_Float16*)hbuf)[i] = ((_Float16*)sh)[i];
    ((float*)cbuf)[i]    = ((float*)sc)[i];
  }
  for (int i = tid; i < BT * NH; i += THREADS) {
    const int row = i >> 6, j = i & 63;
    A[row * KPAD + j] = sh[1][row][j];
  }
  __syncthreads();
  project();
  __syncthreads();
  {
    const int row = tid >> 4, o = tid & 15;   // 256 threads == BT*OUTSZ
    const float v = outst[row][o];
    elem[row][o] = (_Float16)v;
    out[((b0 + row) * TWC + 0) * OUTSZ + o] = v;   // ctx_lens >= 1 always
  }
  __syncthreads();

  // ================= decode over TC-1 steps =================
  for (int s = 0; s < TWC - 1; ++s) {
    for (int idx = tid; idx < BT * KPAD; idx += THREADS) {
      const int row = idx >> 7, col = idx & 127;
      _Float16 v;
      if (col < 16)         v = elem[row][col];
      else if (col < 20)    v = (_Float16)context[((b0 + row) * TWC + s) * 4 + (col - 16)];
      else if (col < 36)    v = (_Float16)__sinf(time_[(b0 + row) * TWC + s] * om[col - 20] + ph[col - 20]);
      else if (col < 64)    v = (_Float16)0.f;
      else                  v = hbuf[0][row][col - 64];
      A[idx] = v;
    }
    __syncthreads();
    gemm_layer(A, gates, biasL[0], wfrag[0], wave, lmod, lhi);
    __syncthreads();
    activate(0, true, true, -1);
    __syncthreads();
    gemm_layer(A, gates, biasL[1], wfrag[1], wave, lmod, lhi);
    __syncthreads();
    activate(1, /*toA_lo=*/true, false, -1);   // h1 -> A cols 0..63 for projection
    __syncthreads();
    project();
    __syncthreads();
    {
      const int row = tid >> 4, o = tid & 15;
      const int tpos = s + 1;
      if (tpos < cl[row]) out[((b0 + row) * TWC + tpos) * OUTSZ + o] = outst[row][o];
    }
    __syncthreads();
  }
}

extern "C" void kernel_launch(void* const* d_in, const int* in_sizes, int n_in,
                              void* d_out, int out_size, void* d_ws, size_t ws_size,
                              hipStream_t stream) {
  const float* x        = (const float*)d_in[0];
  const int*   x_lens   = (const int*)d_in[1];
  const float* time_    = (const float*)d_in[2];
  const float* context  = (const float*)d_in[3];
  const int*   ctx_lens = (const int*)d_in[4];
  const float* omega    = (const float*)d_in[5];
  const float* phi      = (const float*)d_in[6];
  const float* Wih0     = (const float*)d_in[7];
  const float* Whh0     = (const float*)d_in[8];
  const float* bih0     = (const float*)d_in[9];
  const float* bhh0     = (const float*)d_in[10];
  const float* Wih1     = (const float*)d_in[11];
  const float* Whh1     = (const float*)d_in[12];
  const float* bih1     = (const float*)d_in[13];
  const float* bhh1     = (const float*)d_in[14];
  const float* Wd       = (const float*)d_in[15];
  const float* bd       = (const float*)d_in[16];
  float* out = (float*)d_out;

  zero_kernel<<<(out_size + 255) / 256, 256, 0, stream>>>(out, out_size);
  ar_lstm_kernel<<<NB / BT, THREADS, 0, stream>>>(
      x, x_lens, time_, context, ctx_lens, omega, phi,
      Wih0, Whh0, bih0, bhh0, Wih1, Whh1, bih1, bhh1, Wd, bd, out);
}
